// LatencyPredictor_41283225649589
// MI455X (gfx1250) — matrix operand order repaired and checked by reading, not compiled, not run on hardware
//
#include <hip/hip_runtime.h>
#include <hip/hip_bf16.h>

typedef __attribute__((ext_vector_type(16))) _Float16 v16h;
typedef __attribute__((ext_vector_type(8)))  float    v8f;

#define B_N    128
#define L_N    2048
#define T_N    64
#define K1_N   64
#define K2_N   32
#define LP_N   1985                      // L - T + 1
#define TILE_M 32                        // rows of H per workgroup
#define NL_TILES ((LP_N + TILE_M - 1) / TILE_M)   // 63

#define D2SZ   (B_N * LP_N * K2_N)       // 8,130,560
#define RSZ    (B_N * LP_N)              // 254,080
#define STREAM_OUT (D2SZ + 3 * RSZ + 2 * B_N)

// workspace layout
#define WS_W1A_H   0                     // 4096 halves
#define WS_W2A_H   4096                  // 2048 halves
#define WS_W1B_H   6144                  // 4096 halves
#define WS_W2B_H   10240                 // 2048 halves
#define WS_I_OFF_BYTES 32768             // f32 I[s][l][b], 2*LP_N*B_N floats

// ---------------------------------------------------------------------------
// Convert weights to f16 once (WMMA operand precision).
// ---------------------------------------------------------------------------
__global__ void wcvt_kernel(const float* __restrict__ W1a, const float* __restrict__ W2a,
                            const float* __restrict__ W1b, const float* __restrict__ W2b,
                            _Float16* __restrict__ wh) {
    int i = blockIdx.x * blockDim.x + threadIdx.x;
    if (i < 4096) wh[WS_W1A_H + i] = (_Float16)W1a[i];
    if (i < 2048) wh[WS_W2A_H + i] = (_Float16)W2a[i];
    if (i < 4096) wh[WS_W1B_H + i] = (_Float16)W1b[i];
    if (i < 2048) wh[WS_W2B_H + i] = (_Float16)W2b[i];
}

// ---------------------------------------------------------------------------
// Dendrites: Hankel-window GEMM1 (relu) -> GEMM2 (relu) -> row reductions.
// Block = 256 threads (8 waves). Block tile = 32 rows of H for one (s, b).
// ---------------------------------------------------------------------------
__launch_bounds__(256)
__global__ void dend_kernel(const float* __restrict__ xa, const float* __restrict__ xb,
                            const float* __restrict__ b1a, const float* __restrict__ b2a,
                            const float* __restrict__ b1b, const float* __restrict__ b2b,
                            const _Float16* __restrict__ wh,
                            float* __restrict__ Iws, float* __restrict__ out) {
    const int ltile = blockIdx.x;
    const int b     = blockIdx.y;
    const int s     = blockIdx.z;
    const int tid   = threadIdx.x;
    const int wid   = tid >> 5;
    const int lane  = tid & 31;
    const int hi    = lane >> 4;         // half-wave select
    const int ln    = lane & 15;
    const int l0    = ltile * TILE_M;

    const float*     x   = s ? xb  : xa;
    const float*     b1  = s ? b1b : b1a;
    const float*     b2  = s ? b2b : b2a;
    const _Float16*  w1h = wh + (s ? WS_W1B_H : WS_W1A_H);
    const _Float16*  w2h = wh + (s ? WS_W2B_H : WS_W2A_H);
    float*           da  = out + (size_t)s * STREAM_OUT;

    __shared__ _Float16 sx[TILE_M + T_N];            // 95-half sliding window (+pad)
    __shared__ _Float16 sd1[TILE_M][K1_N + 8];       // relu(d1) tile, f16
    __shared__ float    sd2[TILE_M][K2_N + 4];       // relu(d2) tile, f32

    // Stage the x window (Hankel tile needs only 32+63 contiguous values).
    for (int i = tid; i < TILE_M + T_N - 1; i += 256) {
        int g = l0 + i; if (g > L_N - 1) g = L_N - 1;
        sx[i] = (_Float16)x[(size_t)b * L_N + g];
    }
    // Prefetch next tile's window into cache (global_prefetch_b8).
    if (tid < 32 && ltile + 1 < (int)gridDim.x) {
        __builtin_prefetch(&x[(size_t)b * L_N + l0 + TILE_M + tid * 4], 0, 1);
    }

    // GEMM1 wave tiling: 2 M-subtiles x 4 N-subtiles.
    const int mt = wid >> 2;
    const int nt = wid & 3;

    // Preload W1 B-fragments (32x16 f16 layout: K = (v<4?0:16) + hi*8 + (v&3)*2 + h, col = ln).
    v16h bf1[2];
#pragma unroll
    for (int kc = 0; kc < 2; ++kc)
#pragma unroll
        for (int v = 0; v < 8; ++v)
#pragma unroll
            for (int h = 0; h < 2; ++h) {
                int k = kc * 32 + (v < 4 ? 0 : 16) + hi * 8 + (v & 3) * 2 + h;
                bf1[kc][v * 2 + h] = w1h[k * K1_N + nt * 16 + ln];
            }
    __syncthreads();

    // ---- GEMM1: d1 = relu(H @ W1 + b1), K = T = 64 (2 WMMA chunks) ----
    v8f acc = {};
#pragma unroll
    for (int kc = 0; kc < 2; ++kc) {
        v16h af;   // 16-bit A 16x32 layout: t = (v<4?0:16) + hi*8 + (v&3)*2 + h
#pragma unroll
        for (int v = 0; v < 8; ++v)
#pragma unroll
            for (int h = 0; h < 2; ++h) {
                int t = kc * 32 + (v < 4 ? 0 : 16) + hi * 8 + (v & 3) * 2 + h;
                af[v * 2 + h] = sx[mt * 16 + ln + t];   // Hankel: H[m][t] = x[l0+m+t]
            }
        acc = __builtin_amdgcn_wmma_f32_16x16x32_f16(false, af, false, bf1[kc],
                                                     (short)0, acc, false, false);
    }
    {   // bias + relu -> LDS (C/D layout: row = hi*8 + i, col = ln)
        int col = nt * 16 + ln;
        float bias = b1[col];
#pragma unroll
        for (int i = 0; i < 8; ++i) {
            int r = mt * 16 + hi * 8 + i;
            float vv = acc[i] + bias;
            sd1[r][col] = (_Float16)(vv > 0.f ? vv : 0.f);
        }
    }
    __syncthreads();

    // ---- GEMM2: d2 = relu(d1 @ W2 + b2), K1 = 64 (2 chunks), waves 0..3 ----
    if (wid < 4) {
        const int mt2 = wid >> 1, nt2 = wid & 1;
        v8f acc2 = {};
#pragma unroll
        for (int kc = 0; kc < 2; ++kc) {
            v16h af, bf;
#pragma unroll
            for (int v = 0; v < 8; ++v)
#pragma unroll
                for (int h = 0; h < 2; ++h) {
                    int ka = kc * 32 + (v < 4 ? 0 : 16) + hi * 8 + (v & 3) * 2 + h;
                    int kb = kc * 32 + (v < 4 ? 0 : 16) + hi * 8 + (v & 3) * 2 + h;
                    af[v * 2 + h] = sd1[mt2 * 16 + ln][ka];
                    bf[v * 2 + h] = w2h[kb * K2_N + nt2 * 16 + ln];
                }
            acc2 = __builtin_amdgcn_wmma_f32_16x16x32_f16(false, af, false, bf,
                                                          (short)0, acc2, false, false);
        }
        int col2 = nt2 * 16 + ln;
        float bias2 = b2[col2];
#pragma unroll
        for (int i = 0; i < 8; ++i) {
            int r = mt2 * 16 + hi * 8 + i;
            int l = l0 + r;
            float vv = acc2[i] + bias2;
            vv = vv > 0.f ? vv : 0.f;
            sd2[r][col2] = vv;
            if (l < LP_N)
                da[((size_t)b * LP_N + l) * K2_N + col2] = vv;   // d2 output
        }
    }
    __syncthreads();

    // ---- Per-row reductions over K2=32: max / argmax / softmax-soma ----
#pragma unroll
    for (int j = 0; j < 4; ++j) {
        int r = wid + j * 8;             // wave-uniform row
        int l = l0 + r;
        if (l >= LP_N) continue;
        float v = sd2[r][lane];
        float vmax = v; int imax = lane;
#pragma unroll
        for (int off = 16; off; off >>= 1) {
            float ov = __shfl_xor(vmax, off, 32);
            int   oi = __shfl_xor(imax, off, 32);
            if (ov > vmax || (ov == vmax && oi < imax)) { vmax = ov; imax = oi; }
        }
        float e = __expf(v - vmax);
        float se = e, sev = e * v;
#pragma unroll
        for (int off = 16; off; off >>= 1) {
            se  += __shfl_xor(se,  off, 32);
            sev += __shfl_xor(sev, off, 32);
        }
        if (lane == 0) {
            size_t ro = (size_t)b * LP_N + l;
            da[D2SZ + ro]       = (float)imax;      // argmax
            da[D2SZ + RSZ + ro] = vmax;             // max value
            Iws[((size_t)s * LP_N + l) * B_N + b] = sev / se;   // soma_in
        }
    }
}

// ---------------------------------------------------------------------------
// LIF scan: 2 blocks x 128 threads (one sequence per thread), coalesced I
// loads ([l][b] layout) and LDS-transposed spike stores.
// ---------------------------------------------------------------------------
__launch_bounds__(128)
__global__ void lif_kernel(const float* __restrict__ Iws, float* __restrict__ out) {
    const int s    = blockIdx.x;
    const int tid  = threadIdx.x;        // batch index
    const int wid  = tid >> 5;
    const int lane = tid & 31;

    __shared__ float tile[B_N][33];

    float* o  = out + (size_t)s * STREAM_OUT;
    float* sa = o + D2SZ + 2 * RSZ;

    float v = 0.f, ss = 0.f, sst = 0.f;
    int first = LP_N;
    const float decay = 1.f - 1.f / 20.f;    // 1 - 1/TAU

    for (int t0 = 0; t0 < LP_N; t0 += 32) {
        int nt = LP_N - t0; if (nt > 32) nt = 32;
        for (int q = 0; q < nt; ++q) {
            int t = t0 + q;
            float I = Iws[((size_t)s * LP_N + t) * B_N + tid];
            v = v * decay + I;                               // Euler step
            float sp = 1.f / (1.f + __expf(-10.f * (v - 1.f)));  // surrogate spike
            if (v > 1.f && first == LP_N) first = t;         // hard spike latency
            tile[tid][q] = sp;
            ss  += sp;
            sst += sp * (float)t;
            v *= (1.f - sp);                                 // detached reset
        }
        __syncthreads();
        // transpose flush: each wave writes 32 batch rows with contiguous t
        for (int rr = 0; rr < 32; ++rr) {
            int br = wid * 32 + rr;
            if (lane < nt) sa[(size_t)br * LP_N + t0 + lane] = tile[br][lane];
        }
        __syncthreads();
    }
    o[D2SZ + 3 * RSZ + tid]       = (float)first;            // hard latency
    o[D2SZ + 3 * RSZ + B_N + tid] = sst / (ss + 1e-6f);      // soft latency
}

// ---------------------------------------------------------------------------
extern "C" void kernel_launch(void* const* d_in, const int* in_sizes, int n_in,
                              void* d_out, int out_size, void* d_ws, size_t ws_size,
                              hipStream_t stream) {
    const float* xa  = (const float*)d_in[0];
    const float* xb  = (const float*)d_in[1];
    const float* W1a = (const float*)d_in[2];
    const float* b1a = (const float*)d_in[3];
    const float* W2a = (const float*)d_in[4];
    const float* b2a = (const float*)d_in[5];
    const float* W1b = (const float*)d_in[6];
    const float* b1b = (const float*)d_in[7];
    const float* W2b = (const float*)d_in[8];
    const float* b2b = (const float*)d_in[9];

    _Float16* wh  = (_Float16*)d_ws;
    float*    Iws = (float*)((char*)d_ws + WS_I_OFF_BYTES);
    float*    out = (float*)d_out;

    wcvt_kernel<<<16, 256, 0, stream>>>(W1a, W2a, W1b, W2b, wh);

    dim3 grid(NL_TILES, B_N, 2);
    dend_kernel<<<grid, 256, 0, stream>>>(xa, xb, b1a, b2a, b1b, b2b, wh, Iws, out);

    lif_kernel<<<2, 128, 0, stream>>>(Iws, out);
}
